// HadamardProduct_32916629357178
// MI455X (gfx1250) — matrix operand order repaired and checked by reading, not compiled, not run on hardware
//
#include <hip/hip_runtime.h>
#include <hip/hip_bf16.h>

// ---------------------------------------------------------------------------
// MI455X (gfx1250) fused implementation:
//   h1 = relu(x1@W1+b1)            [512,256]  bf16 in ws   (WMMA bf16)
//   h2 = relu(x2@W2+b2)            [4096,256] bf16 in ws   (WMMA bf16)
//   per (b,n): A = LN(h1_row * h2[b])  -> LDS bf16 [256][272]
//              W3^T staged in LDS [256][272] bf16 (B-fragments via ds_load)
//              Y = A @ W3 + b3 ; out = relu(LN(Y))  (WMMA bf16, LN in regs)
// ---------------------------------------------------------------------------

#define EPS 1e-5f

typedef __attribute__((ext_vector_type(16))) __bf16 v16bf;
typedef __attribute__((ext_vector_type(8)))  float  v8f;

union Frag { uint4 q[2]; v16bf v; };  // 32 bytes: 16 bf16 halves

// Native conversions -> v_cvt_pk_bf16_f32 (verified in round-2 asm).
__device__ __forceinline__ __bf16 f2bf(float f) { return (__bf16)f; }
__device__ __forceinline__ unsigned short f2bfu(float f) {
  union { __bf16 b; unsigned short s; } o; o.b = (__bf16)f; return o.s;
}
__device__ __forceinline__ float bfu2f(unsigned short s) {
  union { unsigned short s; __bf16 b; } v; v.s = s; return (float)v.b;
}
__device__ __forceinline__ void unpack8(const uint4 q, float* f) {
  const unsigned u0 = q.x, u1 = q.y, u2 = q.z, u3 = q.w;
  f[0] = bfu2f((unsigned short)(u0 & 0xffffu)); f[1] = bfu2f((unsigned short)(u0 >> 16));
  f[2] = bfu2f((unsigned short)(u1 & 0xffffu)); f[3] = bfu2f((unsigned short)(u1 >> 16));
  f[4] = bfu2f((unsigned short)(u2 & 0xffffu)); f[5] = bfu2f((unsigned short)(u2 >> 16));
  f[6] = bfu2f((unsigned short)(u3 & 0xffffu)); f[7] = bfu2f((unsigned short)(u3 >> 16));
}
__device__ __forceinline__ uint4 pack8(const float* f) {
  uint4 q;
  q.x = (unsigned)f2bfu(f[0]) | ((unsigned)f2bfu(f[1]) << 16);
  q.y = (unsigned)f2bfu(f[2]) | ((unsigned)f2bfu(f[3]) << 16);
  q.z = (unsigned)f2bfu(f[4]) | ((unsigned)f2bfu(f[5]) << 16);
  q.w = (unsigned)f2bfu(f[6]) | ((unsigned)f2bfu(f[7]) << 16);
  return q;
}

// ---------------------------------------------------------------------------
// Kernel 0: W [k][256] f32 -> WT [n][K] bf16  (B-fragment friendly layout).
// ---------------------------------------------------------------------------
__global__ __launch_bounds__(256) void w_transpose_bf16_k(
    const float* __restrict__ W, unsigned short* __restrict__ WT, int K) {
  const int i = blockIdx.x * 256 + threadIdx.x;   // 0..K*256-1
  const int k = i >> 8, n = i & 255;
  WT[(size_t)n * K + k] = f2bfu(W[i]);
}

// ---------------------------------------------------------------------------
// Kernel 1: H = relu(X @ W + bias) -> bf16.  X:[M,K] f32, WT:[256,K] bf16.
// One wave per 16x16 output tile; grid.x = (M/16)*16.
// A-frag per ISA 16-bit A layout: lanes<16 K={0..7,16..23}, lanes>=16 +8.
// B-frag: lane = column, same K order as A-frag (lanes<16 K={0..7,16..23}, lanes>=16 +8).
// ---------------------------------------------------------------------------
__global__ __launch_bounds__(32) void gemm_relu_bf16_k(
    const float* __restrict__ X, const unsigned short* __restrict__ WT,
    const float* __restrict__ bias, unsigned short* __restrict__ H, int K) {
  const int lane = threadIdx.x;
  const int tm = blockIdx.x >> 4;
  const int tn = blockIdx.x & 15;
  const int l15 = lane & 15;
  const bool hi = lane >= 16;

  v8f acc = {};
  const int arow = tm * 16 + l15;
  const int bcol = tn * 16 + l15;

  for (int k0 = 0; k0 < K; k0 += 32) {
    Frag a, b;
    const float4* Xq = (const float4*)(X + (size_t)arow * K + k0 + (hi ? 8 : 0));
    const float4 x0 = Xq[0], x1 = Xq[1];   // halves 0..7  -> K offsets +0..7
    const float4 x2 = Xq[4], x3 = Xq[5];   // halves 8..15 -> K offsets +16..23
    a.v[0] = f2bf(x0.x); a.v[1] = f2bf(x0.y); a.v[2]  = f2bf(x0.z); a.v[3]  = f2bf(x0.w);
    a.v[4] = f2bf(x1.x); a.v[5] = f2bf(x1.y); a.v[6]  = f2bf(x1.z); a.v[7]  = f2bf(x1.w);
    a.v[8] = f2bf(x2.x); a.v[9] = f2bf(x2.y); a.v[10] = f2bf(x2.z); a.v[11] = f2bf(x2.w);
    a.v[12] = f2bf(x3.x); a.v[13] = f2bf(x3.y); a.v[14] = f2bf(x3.z); a.v[15] = f2bf(x3.w);
    const unsigned short* bp = WT + (size_t)bcol * K + k0 + (hi ? 8 : 0);
    b.q[0] = *(const uint4*)(const void*)(bp);
    b.q[1] = *(const uint4*)(const void*)(bp + 16);
    acc = __builtin_amdgcn_wmma_f32_16x16x32_bf16(false, a.v, false, b.v,
                                                  (short)0, acc, false, false);
  }
  const float bv = bias[bcol];
#pragma unroll
  for (int r = 0; r < 8; ++r) {
    const int row = tm * 16 + r + (hi ? 8 : 0);
    float y = acc[r] + bv;
    y = y > 0.f ? y : 0.f;
    H[(size_t)row * 256 + bcol] = f2bfu(y);
  }
}

// ---------------------------------------------------------------------------
// Kernel 2 (fused): one workgroup per (b,n). 256 threads = 8 waves.
//   LDS: As [256][272] bf16 (LN'd Hadamard rows), Bs [256][272] bf16 (W3^T)
//        total 278528 B  (<= 320 KB WGP LDS)
//   Stage A: build As + copy W3^T -> Bs, one barrier.
//   Stage B: each wave: 2 tile-rows x 16 tile-cols, K=256 (8 WMMA k-steps),
//            bias + row-LayerNorm + relu in registers, NT stores to out.
// ---------------------------------------------------------------------------
#define AS_STRIDE 272  // 256 + 16 halves pad; 16/32B aligned, conflict-light

__global__ __launch_bounds__(256) void fused_had_ln_gemm_ln_k(
    const unsigned short* __restrict__ h1bf,   // [512,256]  bf16
    const unsigned short* __restrict__ h2bf,   // [4096,256] bf16
    const unsigned short* __restrict__ w3t,    // [256(n),256(k)] bf16
    const float* __restrict__ b3,
    const float* __restrict__ g0,  const float* __restrict__ be0,
    const float* __restrict__ g1,  const float* __restrict__ be1,
    float* __restrict__ out) {
  extern __shared__ unsigned short lds[];
  unsigned short* As = lds;                    // [256][AS_STRIDE]
  unsigned short* Bs = lds + 256 * AS_STRIDE;  // [256][AS_STRIDE]

  const int bn   = blockIdx.x;                 // 0..511  (b*32 + n)
  const int b    = bn >> 5;
  const int tid  = threadIdx.x;
  const int lane = tid & 31;
  const int wave = tid >> 5;
  const int l15  = lane & 15;
  const bool hi  = lane >= 16;

  // ---- stage W3^T into LDS (128 KB, b128 copies, coalesced) ----
  {
    // 256*32 uint4-chunks of 8 halves; chunk c -> row n = c>>5, j = c&31
    for (int c = tid; c < 256 * 32; c += 256) {
      const int n = c >> 5, j = c & 31;
      const uint4 q = *(const uint4*)(const void*)(w3t + n * 256 + j * 8);
      *(uint4*)(void*)(Bs + n * AS_STRIDE + j * 8) = q;
    }
  }

  // ---------------- Stage A: Hadamard + first LayerNorm ----------------
  float h1v[8], gv[8], bev[8];
  unpack8(*(const uint4*)(const void*)(h1bf + bn * 256 + lane * 8), h1v);
  {
    const float4* gq  = (const float4*)(g0  + lane * 8);
    const float4* beq = (const float4*)(be0 + lane * 8);
    float4 t0 = gq[0],  t1 = gq[1];
    gv[0]=t0.x; gv[1]=t0.y; gv[2]=t0.z; gv[3]=t0.w;
    gv[4]=t1.x; gv[5]=t1.y; gv[6]=t1.z; gv[7]=t1.w;
    t0 = beq[0]; t1 = beq[1];
    bev[0]=t0.x; bev[1]=t0.y; bev[2]=t0.z; bev[3]=t0.w;
    bev[4]=t1.x; bev[5]=t1.y; bev[6]=t1.z; bev[7]=t1.w;
  }
  for (int t = wave * 32; t < wave * 32 + 32; ++t) {
    float h2v[8], p[8];
    unpack8(*(const uint4*)(const void*)(h2bf + (size_t)(b * 256 + t) * 256 + lane * 8), h2v);
    float s = 0.f, s2 = 0.f;
#pragma unroll
    for (int j = 0; j < 8; ++j) {
      const float v = h1v[j] * h2v[j];
      p[j] = v; s += v; s2 += v * v;
    }
#pragma unroll
    for (int m = 16; m >= 1; m >>= 1) {
      s  += __shfl_xor(s,  m, 32);
      s2 += __shfl_xor(s2, m, 32);
    }
    const float mu  = s * (1.f / 256.f);
    const float var = s2 * (1.f / 256.f) - mu * mu;
    const float rs  = rsqrtf(var + EPS);
    float yv[8];
#pragma unroll
    for (int j = 0; j < 8; ++j) yv[j] = (p[j] - mu) * rs * gv[j] + bev[j];
    *(uint4*)(void*)(As + t * AS_STRIDE + lane * 8) = pack8(yv);
  }
  __syncthreads();

  // ---------------- Stage B: WMMA GEMM + bias + LN + relu ----------------
  float* obase = out + (size_t)bn * 256 * 256;
  float b3v[16], g1v[16], be1v[16];
#pragma unroll
  for (int tn = 0; tn < 16; ++tn) {
    const int c = tn * 16 + l15;
    b3v[tn] = b3[c]; g1v[tn] = g1[c]; be1v[tn] = be1[c];
  }

  for (int tmi = 0; tmi < 2; ++tmi) {
    const int tm = wave + 8 * tmi;               // tile-row (T dim)
    const v8f zero = {};
    v8f acc[16];
#pragma unroll
    for (int tn = 0; tn < 16; ++tn) acc[tn] = zero;

    for (int kk = 0; kk < 8; ++kk) {             // K = 8 * 32
      Frag a;
      const unsigned short* ap =
          As + (tm * 16 + l15) * AS_STRIDE + kk * 32 + (hi ? 8 : 0);
      a.q[0] = *(const uint4*)(const void*)(ap);
      a.q[1] = *(const uint4*)(const void*)(ap + 16);
#pragma unroll
      for (int tn = 0; tn < 16; ++tn) {
        Frag bb;
        const unsigned short* bp =
            Bs + (tn * 16 + l15) * AS_STRIDE + kk * 32 + (hi ? 8 : 0);
        bb.q[0] = *(const uint4*)(const void*)(bp);
        bb.q[1] = *(const uint4*)(const void*)(bp + 16);
        acc[tn] = __builtin_amdgcn_wmma_f32_16x16x32_bf16(
            false, a.v, false, bb.v, (short)0, acc[tn], false, false);
      }
    }

    // bias, then per-row stats (row lives in one 16-lane half-group)
    float s[8], s2[8];
#pragma unroll
    for (int r = 0; r < 8; ++r) { s[r] = 0.f; s2[r] = 0.f; }
#pragma unroll
    for (int tn = 0; tn < 16; ++tn) {
#pragma unroll
      for (int r = 0; r < 8; ++r) {
        const float y = acc[tn][r] + b3v[tn];
        acc[tn][r] = y;
        s[r] += y; s2[r] += y * y;
      }
    }
#pragma unroll
    for (int m = 8; m >= 1; m >>= 1) {
#pragma unroll
      for (int r = 0; r < 8; ++r) {
        s[r]  += __shfl_xor(s[r],  m, 32);   // stays inside 16-lane half
        s2[r] += __shfl_xor(s2[r], m, 32);
      }
    }
#pragma unroll
    for (int r = 0; r < 8; ++r) {
      const float mu  = s[r] * (1.f / 256.f);
      const float var = s2[r] * (1.f / 256.f) - mu * mu;
      const float rs  = rsqrtf(var + EPS);
      const int t = tm * 16 + r + (hi ? 8 : 0);
      float* orow = obase + (size_t)t * 256;
#pragma unroll
      for (int tn = 0; tn < 16; ++tn) {
        float y = (acc[tn][r] - mu) * rs * g1v[tn] + be1v[tn];
        y = y > 0.f ? y : 0.f;
        __builtin_nontemporal_store(y, orow + tn * 16 + l15);
      }
    }
  }
}

// ---------------------------------------------------------------------------
extern "C" void kernel_launch(void* const* d_in, const int* in_sizes, int n_in,
                              void* d_out, int out_size, void* d_ws, size_t ws_size,
                              hipStream_t stream) {
  (void)in_sizes; (void)n_in; (void)out_size; (void)ws_size;
  const float* x1  = (const float*)d_in[0];
  const float* x2  = (const float*)d_in[1];
  const float* W1  = (const float*)d_in[2];
  const float* b1  = (const float*)d_in[3];
  const float* W2  = (const float*)d_in[4];
  const float* b2  = (const float*)d_in[5];
  const float* W3  = (const float*)d_in[6];
  const float* b3  = (const float*)d_in[7];
  const float* g0  = (const float*)d_in[8];
  const float* be0 = (const float*)d_in[9];
  const float* g1  = (const float*)d_in[10];
  const float* be1 = (const float*)d_in[11];

  unsigned short* h1bf = (unsigned short*)d_ws;          // 512*256    bf16
  unsigned short* h2bf = h1bf + 512 * 256;               // 4096*256   bf16
  unsigned short* w3t  = h2bf + 4096 * 256;              // 256*256    bf16
  unsigned short* w1t  = w3t + 256 * 256;                // 256*512    bf16
  unsigned short* w2t  = w1t + 256 * 512;                // 256*512    bf16

  // Pre-transpose + convert weights to bf16 (tiny, one-time).
  w_transpose_bf16_k<<<512, 256, 0, stream>>>(W1, w1t, 512);
  w_transpose_bf16_k<<<512, 256, 0, stream>>>(W2, w2t, 512);
  w_transpose_bf16_k<<<256, 256, 0, stream>>>(W3, w3t, 256);

  // h1 = relu(x1@W1+b1): M=512 -> 32x16 tiles
  gemm_relu_bf16_k<<<512, 32, 0, stream>>>(x1, w1t, b1, h1bf, 512);
  // h2 = relu(x2@W2+b2): M=4096 -> 256x16 tiles
  gemm_relu_bf16_k<<<4096, 32, 0, stream>>>(x2, w2t, b2, h2bf, 512);

  // fused Hadamard + LN + GEMM + LN + relu : one WG per (b,n)
  const size_t lds_bytes = 2u * 256 * AS_STRIDE * sizeof(unsigned short); // 278528
  fused_had_ln_gemm_ln_k<<<512, 256, lds_bytes, stream>>>(
      h1bf, h2bf, w3t, b3, g0, be0, g1, be1, (float*)d_out);
}
